// RateMatrixOutputHead_1468878815880
// MI455X (gfx1250) — matrix operand order repaired and checked by reading, not compiled, not run on hardware
//
#include <hip/hip_runtime.h>
#include <math.h>

// ---------------------------------------------------------------------------
// Problem constants (reference: B,L,E,V = 4,2048,1024,64; NT = 2016)
// ---------------------------------------------------------------------------
#define BATCH  4
#define SEQL   2048
#define EDIM   1024
#define VDIM   64
#define NTRI   2016
#define MROWS  (BATCH * SEQL)        // 8192
#define LN_EPS 1e-5f

typedef __attribute__((ext_vector_type(16))) __bf16        v16bf;
typedef __attribute__((ext_vector_type(8)))  float         v8f;
typedef __attribute__((ext_vector_type(4)))  unsigned int  u32x4;

union FragU {
    v16bf v;
    u32x4 u[2];
};

// round-to-nearest-even f32 -> bf16 (raw u16)
static __device__ __forceinline__ unsigned short f32_to_bf16(float f) {
    unsigned int u = __float_as_uint(f);
    u += 0x7FFFu + ((u >> 16) & 1u);
    return (unsigned short)(u >> 16);
}

// LDS byte offset of a shared-memory pointer (generic addr low 32 bits)
static __device__ __forceinline__ unsigned int lds_offset(const void* p) {
    return (unsigned int)(unsigned long long)p;
}

// ---------------------------------------------------------------------------
// Kernel: f32 -> bf16 conversion (weights only; grid-stride)
// ---------------------------------------------------------------------------
__global__ __launch_bounds__(256) void cvt_bf16_kernel(const float* __restrict__ src,
                                                       unsigned short* __restrict__ dst,
                                                       int n) {
    int i = blockIdx.x * blockDim.x + threadIdx.x;
    int stride = gridDim.x * blockDim.x;
    for (; i < n; i += stride) dst[i] = f32_to_bf16(src[i]);
}

// ---------------------------------------------------------------------------
// Fused: dense (8192x1024x1024 WMMA bf16) + bias + exact GELU + LayerNorm.
// One block = 16-row M-tile x full N=1024. A tile staged to LDS (f32->bf16
// fused). GELU output lives in LDS; LN via wave32 shuffles; bf16 out only.
// ---------------------------------------------------------------------------
__global__ __launch_bounds__(256) void dense_gelu_ln_kernel(
        const float* __restrict__ hx,             // [M,E] f32
        const unsigned short* __restrict__ Wd,    // [E,E] bf16
        const float* __restrict__ bd,             // [E]
        const float* __restrict__ lng, const float* __restrict__ lnb,
        unsigned short* __restrict__ Hbf) {       // [M,E] bf16 out
    __shared__ unsigned short Ash[16 * EDIM] __attribute__((aligned(16)));  // 32 KB
    __shared__ float Hs[16 * EDIM];                                         // 64 KB
    __shared__ float mu_s[16];
    __shared__ float inv_s[16];

    const int t    = threadIdx.x;
    const int wave = t >> 5;
    const int lane = t & 31;
    const int m0   = blockIdx.x * 16;

    // ---- stage A: 16 contiguous rows of hx, convert f32 -> bf16 into LDS
    {
        const float4* src = (const float4*)(hx + (size_t)m0 * EDIM);
        unsigned int* dst = (unsigned int*)Ash;
        for (int i = t; i < (16 * EDIM) / 4; i += 256) {
            float4 v = src[i];
            dst[i * 2 + 0] = (unsigned)f32_to_bf16(v.x) | ((unsigned)f32_to_bf16(v.y) << 16);
            dst[i * 2 + 1] = (unsigned)f32_to_bf16(v.z) | ((unsigned)f32_to_bf16(v.w) << 16);
        }
    }
    __syncthreads();

    // ---- WMMA GEMM: each wave does 4 double-tiles (32 cols each) of N=1024
    const int row = lane & 15;
    const int hi  = lane >> 4;
    const int klo = hi * 16;
    const int kbh = hi * 16;
    const unsigned short* Arow = Ash + row * EDIM;

    for (int q = 0; q < 4; ++q) {
        const int n0 = (wave + 8 * q) * 32;
        const unsigned short* B0 = Wd + (size_t)(n0 + row) * EDIM + kbh;
        const unsigned short* B1 = Wd + (size_t)(n0 + 16 + row) * EDIM + kbh;

        v8f acc0 = {};
        v8f acc1 = {};
#pragma unroll 4
        for (int k0 = 0; k0 < EDIM; k0 += 32) {
            FragU a, b0, b1;
            const u32x4* ap = (const u32x4*)(Arow + k0 + klo);   // ds_load_b128
            a.u[0] = ap[0];
            a.u[1] = ap[1];
            const u32x4* bp0 = (const u32x4*)(B0 + k0);
            b0.u[0] = bp0[0];
            b0.u[1] = bp0[1];
            const u32x4* bp1 = (const u32x4*)(B1 + k0);
            b1.u[0] = bp1[0];
            b1.u[1] = bp1[1];
            acc0 = __builtin_amdgcn_wmma_f32_16x16x32_bf16(false, a.v, false, b0.v,
                                                           (short)0, acc0, false, false);
            acc1 = __builtin_amdgcn_wmma_f32_16x16x32_bf16(false, a.v, false, b1.v,
                                                           (short)0, acc1, false, false);
        }
        const int nA = n0 + row;
        const int nB = n0 + 16 + row;
        const float biasA = bd[nA];
        const float biasB = bd[nB];
#pragma unroll
        for (int r = 0; r < 8; ++r) {
            const int mloc = r + hi * 8;
            float x0 = acc0[r] + biasA;
            float x1 = acc1[r] + biasB;
            x0 = 0.5f * x0 * (1.0f + erff(x0 * 0.70710678118654752f));
            x1 = 0.5f * x1 * (1.0f + erff(x1 * 0.70710678118654752f));
            Hs[mloc * EDIM + nA] = x0;
            Hs[mloc * EDIM + nB] = x1;
        }
    }
    __syncthreads();

    // ---- LayerNorm: 16 threads per row, shuffle reductions within 16-groups
    {
        const int r   = t >> 4;
        const int l16 = t & 15;
        const float* hr = Hs + r * EDIM;
        float s = 0.0f;
#pragma unroll 8
        for (int j = 0; j < 64; ++j) s += hr[l16 + 16 * j];
#pragma unroll
        for (int msk = 1; msk < 16; msk <<= 1) s += __shfl_xor(s, msk);
        const float mu = s * (1.0f / EDIM);

        float vs = 0.0f;
#pragma unroll 8
        for (int j = 0; j < 64; ++j) {
            float d = hr[l16 + 16 * j] - mu;
            vs += d * d;
        }
#pragma unroll
        for (int msk = 1; msk < 16; msk <<= 1) vs += __shfl_xor(vs, msk);
        if (l16 == 0) {
            mu_s[r]  = mu;
            inv_s[r] = rsqrtf(vs * (1.0f / EDIM) + LN_EPS);
        }
    }
    __syncthreads();

    // ---- normalize + cast + coalesced bf16 store (2 elems per u32)
    {
        unsigned int* outp = (unsigned int*)(Hbf + (size_t)m0 * EDIM);
        for (int p = t; p < 16 * EDIM / 2; p += 256) {
            const int r = p >> 9;
            const int c = (p & 511) * 2;
            const float mu = mu_s[r], inv = inv_s[r];
            float a = (Hs[r * EDIM + c]     - mu) * inv * lng[c]     + lnb[c];
            float b = (Hs[r * EDIM + c + 1] - mu) * inv * lng[c + 1] + lnb[c + 1];
            outp[p] = (unsigned)f32_to_bf16(a) | ((unsigned)f32_to_bf16(b) << 16);
        }
    }
}

// ---------------------------------------------------------------------------
// Generic WMMA bf16 GEMM (used for theta logits, N=64): bias epilogue.
// One wave -> 16x32 tile; grid sized so wave count == tile count.
// ---------------------------------------------------------------------------
__global__ __launch_bounds__(256) void gemm_bias_kernel(
        const unsigned short* __restrict__ A, const unsigned short* __restrict__ W,
        const float* __restrict__ bias, float* __restrict__ C,
        int N, int K) {
    const int wave = threadIdx.x >> 5;
    const int lane = threadIdx.x & 31;
    const int ntiles2 = N >> 5;
    const int gtile = blockIdx.x * 8 + wave;
    const int m0 = (gtile / ntiles2) * 16;
    const int n0 = (gtile % ntiles2) * 32;

    const int row = lane & 15;
    const int hi  = lane >> 4;
    const unsigned short* Arow = A + (size_t)(m0 + row) * K;
    const unsigned short* B0  = W + (size_t)(n0 + row) * K + hi * 16;
    const unsigned short* B1  = W + (size_t)(n0 + 16 + row) * K + hi * 16;

    v8f acc0 = {};
    v8f acc1 = {};
#pragma unroll 4
    for (int k0 = 0; k0 < K; k0 += 32) {
        FragU a, b0, b1;
        const u32x4* ap = (const u32x4*)(Arow + k0 + hi * 16);
        a.u[0] = ap[0];
        a.u[1] = ap[1];
        const u32x4* bp0 = (const u32x4*)(B0 + k0);
        b0.u[0] = bp0[0];
        b0.u[1] = bp0[1];
        const u32x4* bp1 = (const u32x4*)(B1 + k0);
        b1.u[0] = bp1[0];
        b1.u[1] = bp1[1];
        acc0 = __builtin_amdgcn_wmma_f32_16x16x32_bf16(false, a.v, false, b0.v,
                                                       (short)0, acc0, false, false);
        acc1 = __builtin_amdgcn_wmma_f32_16x16x32_bf16(false, a.v, false, b1.v,
                                                       (short)0, acc1, false, false);
    }
    const int nA = n0 + row, nB = n0 + 16 + row;
    const float biasA = bias[nA], biasB = bias[nB];
#pragma unroll
    for (int r = 0; r < 8; ++r) {
        const int mr = m0 + r + hi * 8;
        C[(size_t)mr * N + nA] = acc0[r] + biasA;
        C[(size_t)mr * N + nB] = acc1[r] + biasB;
    }
}

// ---------------------------------------------------------------------------
// Softmax over V=64: one wave per row, 2 elements per lane (wave32 shuffles).
// ---------------------------------------------------------------------------
__global__ __launch_bounds__(256) void softmax_kernel(const float* __restrict__ L,
                                                      float* __restrict__ Pi) {
    const int wave = threadIdx.x >> 5;
    const int lane = threadIdx.x & 31;
    const int r    = blockIdx.x * 8 + wave;
    const float* lr = L + (size_t)r * VDIM;

    float x0 = lr[lane];
    float x1 = lr[lane + 32];
    float mx = fmaxf(x0, x1);
#pragma unroll
    for (int off = 16; off > 0; off >>= 1) mx = fmaxf(mx, __shfl_xor(mx, off));
    float e0 = expf(x0 - mx);
    float e1 = expf(x1 - mx);
    float s = e0 + e1;
#pragma unroll
    for (int off = 16; off > 0; off >>= 1) s += __shfl_xor(s, off);
    const float inv = 1.0f / s;
    Pi[(size_t)r * VDIM + lane]      = e0 * inv;
    Pi[(size_t)r * VDIM + lane + 32] = e1 * inv;
}

// ---------------------------------------------------------------------------
// Fused: Theta GEMM (8192x2016x1024 WMMA bf16) + softplus + Q assembly.
// One block = 16-row M-tile. A tile staged via ASYNC global->LDS copy
// (global_load_async_to_lds_b128 / s_wait_asynccnt). Theta kept in LDS
// (129 KB) -> the 132 MB HBM round-trip for Theta is eliminated.
// ---------------------------------------------------------------------------
__global__ __launch_bounds__(256) void theta_q_kernel(
        const unsigned short* __restrict__ Hbf,   // [M,E] bf16
        const unsigned short* __restrict__ Wt,    // [NT,E] bf16
        const float* __restrict__ bTh,            // [NT]
        const float* __restrict__ Pi,             // [M,V]
        float* __restrict__ Q) {                  // [M,V,V]
    __shared__ unsigned short Ash[16 * EDIM] __attribute__((aligned(16)));  // 32 KB
    __shared__ float ths[16][NTRI];                                         // 129 KB
    __shared__ float sqp[16][VDIM];                                         // 4 KB

    const int t    = threadIdx.x;
    const int wave = t >> 5;
    const int lane = t & 31;
    const int m0   = blockIdx.x * 16;

    // ---- async stage of the contiguous 32 KB A tile (ASYNCcnt path)
    {
        const unsigned int lbase = lds_offset(&Ash[0]);
        const unsigned long long gbase =
            (unsigned long long)(Hbf + (size_t)m0 * EDIM);
#pragma unroll
        for (int i = 0; i < 8; ++i) {
            const int c = t + i * 256;                 // 16-byte chunk index
            unsigned int       loff = lbase + (unsigned)c * 16u;
            unsigned long long ga   = gbase + (unsigned long long)c * 16ull;
            asm volatile("global_load_async_to_lds_b128 %0, %1, off"
                         :: "v"(loff), "v"(ga) : "memory");
        }
    }
    // overlap: stage sqrt(pi) while the async copy is in flight
    for (int i = t; i < 16 * VDIM; i += 256)
        (&sqp[0][0])[i] = sqrtf(Pi[(size_t)m0 * VDIM + i]);
    asm volatile("s_wait_asynccnt 0x0" ::: "memory");
    __syncthreads();

    // ---- WMMA GEMM over all 63 double-tiles of N=2016, A from LDS
    const int row = lane & 15;
    const int hi  = lane >> 4;
    const unsigned short* Arow = Ash + row * EDIM;

    for (int dt = wave; dt < NTRI / 32; dt += 8) {
        const int n0 = dt * 32;
        const unsigned short* B0 = Wt + (size_t)(n0 + row) * EDIM + hi * 16;
        const unsigned short* B1 = Wt + (size_t)(n0 + 16 + row) * EDIM + hi * 16;

        v8f acc0 = {};
        v8f acc1 = {};
#pragma unroll 4
        for (int k0 = 0; k0 < EDIM; k0 += 32) {
            FragU a, b0, b1;
            const u32x4* ap = (const u32x4*)(Arow + k0 + hi * 16);  // ds_load_b128
            a.u[0] = ap[0];
            a.u[1] = ap[1];
            const u32x4* bp0 = (const u32x4*)(B0 + k0);
            b0.u[0] = bp0[0];
            b0.u[1] = bp0[1];
            const u32x4* bp1 = (const u32x4*)(B1 + k0);
            b1.u[0] = bp1[0];
            b1.u[1] = bp1[1];
            acc0 = __builtin_amdgcn_wmma_f32_16x16x32_bf16(false, a.v, false, b0.v,
                                                           (short)0, acc0, false, false);
            acc1 = __builtin_amdgcn_wmma_f32_16x16x32_bf16(false, a.v, false, b1.v,
                                                           (short)0, acc1, false, false);
        }
        const int nA = n0 + row, nB = n0 + 16 + row;
        const float biasA = bTh[nA], biasB = bTh[nB];
#pragma unroll
        for (int r = 0; r < 8; ++r) {
            const int mloc = r + hi * 8;
            float x0 = acc0[r] + biasA;
            float x1 = acc1[r] + biasB;
            x0 = (x0 > 20.0f) ? x0 : log1pf(expf(x0));   // softplus
            x1 = (x1 > 20.0f) ? x1 : log1pf(expf(x1));
            ths[mloc][nA] = x0;
            ths[mloc][nB] = x1;
        }
    }
    __syncthreads();

    // ---- Q assembly for the 16 rows, straight from LDS
    const int i  = t >> 2;     // Q row 0..63
    const int jb = t & 3;      // 16-wide column block
    for (int m = 0; m < 16; ++m) {
        const float inv_si = 1.0f / sqp[m][i];
        float* qrow = Q + ((size_t)(m0 + m) * VDIM + i) * VDIM;
        float partial = 0.0f;
#pragma unroll
        for (int blk = 0; blk < 4; ++blk) {
            float4 qv;
            float tmp[4];
#pragma unroll
            for (int e = 0; e < 4; ++e) {
                const int j = jb * 16 + blk * 4 + e;
                float q = 0.0f;
                if (j != i) {
                    const int a = (i < j) ? i : j;
                    const int c = (i < j) ? j : i;
                    const int idx = (a * (2 * VDIM - a - 1)) / 2 + (c - a - 1);
                    q = ths[m][idx] * sqp[m][j] * inv_si;
                }
                partial += q;
                tmp[e] = q;
            }
            qv.x = tmp[0]; qv.y = tmp[1]; qv.z = tmp[2]; qv.w = tmp[3];
            ((float4*)qrow)[jb * 4 + blk] = qv;
        }
        // 4 adjacent lanes share one (m,i) row: reduce, then the lane that
        // owns the diagonal's float4 overwrites it (same-thread ordering).
        partial += __shfl_xor(partial, 1);
        partial += __shfl_xor(partial, 2);
        if (jb == (i >> 4)) qrow[i] = -partial;
    }
}

// ---------------------------------------------------------------------------
// Host-side launcher
// ---------------------------------------------------------------------------
extern "C" void kernel_launch(void* const* d_in, const int* in_sizes, int n_in,
                              void* d_out, int out_size, void* d_ws, size_t ws_size,
                              hipStream_t stream) {
    (void)in_sizes; (void)n_in; (void)out_size; (void)ws_size;

    const float* hx  = (const float*)d_in[0];  // (B,L,E)
    const float* Wd  = (const float*)d_in[1];  // (E,E)
    const float* bd  = (const float*)d_in[2];  // (E)
    const float* lng = (const float*)d_in[3];  // (E)
    const float* lnb = (const float*)d_in[4];  // (E)
    const float* Wth = (const float*)d_in[5];  // (V,E)
    const float* bth = (const float*)d_in[6];  // (V)
    const float* WTh = (const float*)d_in[7];  // (NT,E)
    const float* bTh = (const float*)d_in[8];  // (NT)

    float* Qout  = (float*)d_out;                              // (B,L,V,V)
    float* PiOut = Qout + (size_t)MROWS * VDIM * VDIM;         // (B,L,V)

    // workspace layout (bytes): weights bf16 + h bf16 + logits f32
    char* ws = (char*)d_ws;
    const size_t OFF_WDBF  = 0;                                      //  2.10 MB
    const size_t OFF_WTHBF = OFF_WDBF  + (size_t)EDIM * EDIM * 2;    //  0.13 MB
    const size_t OFF_WTBF  = OFF_WTHBF + (size_t)VDIM * EDIM * 2;    //  4.13 MB
    const size_t OFF_HBF   = OFF_WTBF  + (size_t)NTRI * EDIM * 2;    // 16.78 MB
    const size_t OFF_LOG   = OFF_HBF   + (size_t)MROWS * EDIM * 2;   //  2.10 MB

    unsigned short* Wd_bf  = (unsigned short*)(ws + OFF_WDBF);
    unsigned short* Wth_bf = (unsigned short*)(ws + OFF_WTHBF);
    unsigned short* WTh_bf = (unsigned short*)(ws + OFF_WTBF);
    unsigned short* h_bf   = (unsigned short*)(ws + OFF_HBF);
    float*          logits = (float*)(ws + OFF_LOG);

    // 1) weight casts to bf16 (hx cast is fused into the dense kernel)
    cvt_bf16_kernel<<<1024, 256, 0, stream>>>(Wd,  Wd_bf,  EDIM * EDIM);
    cvt_bf16_kernel<<<256,  256, 0, stream>>>(Wth, Wth_bf, VDIM * EDIM);
    cvt_bf16_kernel<<<1024, 256, 0, stream>>>(WTh, WTh_bf, NTRI * EDIM);

    // 2) fused dense + GELU + LayerNorm -> bf16 activations (512 blocks)
    dense_gelu_ln_kernel<<<MROWS / 16, 256, 0, stream>>>(hx, Wd_bf, bd, lng, lnb,
                                                         h_bf);

    // 3) theta logits (8192x64x1024): 1024 wave-tiles -> 128 blocks
    gemm_bias_kernel<<<128, 256, 0, stream>>>(h_bf, Wth_bf, bth, logits,
                                              VDIM, EDIM);

    // 4) softmax -> pi
    softmax_kernel<<<MROWS / 8, 256, 0, stream>>>(logits, PiOut);

    // 5) fused Theta GEMM + softplus + Q assembly (512 blocks)
    theta_q_kernel<<<MROWS / 16, 256, 0, stream>>>(h_bf, WTh_bf, bTh, PiOut, Qout);
}
